// GCN_10299331576451
// MI455X (gfx1250) — matrix operand order repaired and checked by reading, compiled, not run on hardware
//
#include <hip/hip_runtime.h>
#include <hip/hip_bf16.h>

// ---------------------------------------------------------------------------
// Types for WMMA (gfx1250, wave32)
// ---------------------------------------------------------------------------
typedef __attribute__((ext_vector_type(16))) _Float16 v16h;
typedef __attribute__((ext_vector_type(8)))  float    v8f;
typedef __attribute__((ext_vector_type(4)))  unsigned int v4u;

union Frag16 {
    v16h h;
    v4u  u[2];
};

#define NODES_PER_GEMM_BLOCK 64
#define LDS_STRIDE 136   // 128 + 8 halves pad; 272B rows keep 16B alignment

// ---------------------------------------------------------------------------
// Wave reductions (wave32)
// ---------------------------------------------------------------------------
__device__ __forceinline__ float wave_max32(float v) {
    for (int m = 16; m > 0; m >>= 1) v = fmaxf(v, __shfl_xor(v, m, 32));
    return v;
}
__device__ __forceinline__ float wave_sum32(float v) {
    for (int m = 16; m > 0; m >>= 1) v += __shfl_xor(v, m, 32);
    return v;
}

// ---------------------------------------------------------------------------
// Degree / normalization
// ---------------------------------------------------------------------------
__global__ void k_fill1(float* deg, int n) {
    int i = blockIdx.x * blockDim.x + threadIdx.x;
    if (i < n) deg[i] = 1.0f;  // self-loop contributes 1
}

__global__ void k_deg(const int* __restrict__ dst, float* deg, int e) {
    int i = blockIdx.x * blockDim.x + threadIdx.x;
    if (i < e) atomicAdd(&deg[dst[i]], 1.0f);
}

__global__ void k_rsqrt(float* dis, int n) {
    int i = blockIdx.x * blockDim.x + threadIdx.x;
    if (i < n) dis[i] = rsqrtf(dis[i]);   // deg >= 1 always (self loop)
}

// ---------------------------------------------------------------------------
// WMMA GEMM: m[N,128] = A[N,128] @ W[128,128]   (f32 in, f16 WMMA, f32 acc)
// Block: 256 threads (8 waves), 64 rows of A per block.
// Wave w: row tile r = w&3 (16 rows), column group cg = w>>2 (4 tiles of 16).
// ---------------------------------------------------------------------------
__global__ void __launch_bounds__(256)
k_gemm128(const float* __restrict__ A, const float* __restrict__ W,
          float* __restrict__ M, int nrows) {
    __shared__ _Float16 sW[128 * LDS_STRIDE];
    __shared__ _Float16 sA[NODES_PER_GEMM_BLOCK * LDS_STRIDE];

    const int tid      = threadIdx.x;
    const int blockRow = blockIdx.x * NODES_PER_GEMM_BLOCK;

    // Stage W (128x128) as f16: row = K, col = N
    for (int i = tid; i < 128 * 128; i += 256) {
        int k = i >> 7, n = i & 127;
        sW[k * LDS_STRIDE + n] = (_Float16)W[i];
    }
    // Stage A block (64x128) as f16, zero-pad past nrows
    for (int i = tid; i < NODES_PER_GEMM_BLOCK * 128; i += 256) {
        int r = i >> 7, k = i & 127;
        int g = blockRow + r;
        sA[r * LDS_STRIDE + k] = (g < nrows) ? (_Float16)A[(size_t)g * 128 + k]
                                             : (_Float16)0.0f;
    }
    __syncthreads();

    const int wv   = tid >> 5;
    const int lane = tid & 31;
    const int r    = wv & 3;     // row tile within block
    const int cg   = wv >> 2;    // column group: 0 -> N 0..63, 1 -> N 64..127

    v8f acc[4] = {};

    const int half = lane >> 4;   // A K-interleave select
    const int mr   = lane & 15;   // A row within tile

    for (int kc = 0; kc < 4; ++kc) {            // K chunks of 32
        // A fragment: lane=M row, elems 0..7 -> K = kc*32 + 8*half + (0..7),
        //                         elems 8..15 -> K = kc*32 + 16 + 8*half + (0..7)
        Frag16 a;
        const _Float16* ap = &sA[(r * 16 + mr) * LDS_STRIDE + kc * 32 + 8 * half];
        a.u[0] = *(const v4u*)ap;
        a.u[1] = *(const v4u*)(ap + 16);

        // B fragment base: lane = N column (lane%16), elems 0..7 -> K = kc*32 + 8*half + (0..7),
        //                  elems 8..15 -> K = kc*32 + 16 + 8*half + (0..7)
        const _Float16* bp = &sW[(kc * 32 + 8 * half) * LDS_STRIDE + mr];

        for (int c = 0; c < 4; ++c) {
            Frag16 b;
            const _Float16* bq = bp + cg * 64 + c * 16;
            #pragma unroll
            for (int s = 0; s < 8; ++s) {
                b.h[s]     = bq[s * LDS_STRIDE];
                b.h[8 + s] = bq[(16 + s) * LDS_STRIDE];
            }
            acc[c] = __builtin_amdgcn_wmma_f32_16x16x32_f16(
                false, a.h, false, b.h, (short)0, acc[c], false, false);
        }
    }

    // D layout: lane L -> N = L%16; VGPR i -> M = i + 8*(L/16)
    const int nlo = lane & 15;
    const int sel = lane >> 4;
    for (int c = 0; c < 4; ++c) {
        int col = cg * 64 + c * 16 + nlo;
        #pragma unroll
        for (int i = 0; i < 8; ++i) {
            int row = blockRow + r * 16 + sel * 8 + i;
            if (row < nrows) M[(size_t)row * 128 + col] = acc[c][i];
        }
    }
}

// ---------------------------------------------------------------------------
// Aggregation: agg = selfloop + scatter_add over edges
// ---------------------------------------------------------------------------
__global__ void k_selfloop(const float* __restrict__ m, const float* __restrict__ dis,
                           float* __restrict__ agg, int n) {
    int i = blockIdx.x * blockDim.x + threadIdx.x;
    if (i < n * 128) {
        int node = i >> 7;
        float d = dis[node];
        agg[i] = m[i] * d * d;
    }
}

__global__ void k_edges(const int* __restrict__ src, const int* __restrict__ dst,
                        const float* __restrict__ dis, const float* __restrict__ m,
                        float* __restrict__ agg, int e) {
    int gw   = (blockIdx.x * blockDim.x + threadIdx.x) >> 5;  // one wave per edge
    int lane = threadIdx.x & 31;
    if (gw >= e) return;
    int s = src[gw], d = dst[gw];
    float w = dis[s] * dis[d];
    float4 v = ((const float4*)(m + (size_t)s * 128))[lane];
    float* ad = agg + (size_t)d * 128 + lane * 4;
    atomicAdd(ad + 0, v.x * w);
    atomicAdd(ad + 1, v.y * w);
    atomicAdd(ad + 2, v.z * w);
    atomicAdd(ad + 3, v.w * w);
}

__global__ void k_bias_lrelu(float* __restrict__ h, const float* __restrict__ b, int n) {
    int i = blockIdx.x * blockDim.x + threadIdx.x;
    if (i < n * 128) {
        float v = h[i] + b[i & 127];
        h[i] = (v > 0.0f) ? v : 0.01f * v;
    }
}

// ---------------------------------------------------------------------------
// concepts = softmax(h)/max(softmax(h)) = exp(h - rowmax(h))   (one wave/node)
// ---------------------------------------------------------------------------
__global__ void k_concepts(const float* __restrict__ h, float* __restrict__ out, int n) {
    int node = blockIdx.x * 8 + (threadIdx.x >> 5);
    int lane = threadIdx.x & 31;
    if (node >= n) return;
    float4 v = ((const float4*)(h + (size_t)node * 128))[lane];
    float mx = fmaxf(fmaxf(v.x, v.y), fmaxf(v.z, v.w));
    mx = wave_max32(mx);
    float4 c;
    c.x = expf(v.x - mx); c.y = expf(v.y - mx);
    c.z = expf(v.z - mx); c.w = expf(v.w - mx);
    ((float4*)(out + (size_t)node * 128))[lane] = c;
}

// ---------------------------------------------------------------------------
// wcomb[c,d] = alpha_norm[c,d] * lens_w[c,0,d]
//            = exp(|w|/T - max_d |w|/T) * w        (normalizer cancels)
// ---------------------------------------------------------------------------
__global__ void k_wcomb(const float* __restrict__ lensW, float* __restrict__ wcomb) {
    __shared__ float red[128];
    int d = threadIdx.x;
    const float invT = 1.0f / 0.6f;
    for (int c = 0; c < 10; ++c) {
        float w = lensW[c * 128 + d];
        float g = fabsf(w) * invT;
        red[d] = g;
        __syncthreads();
        for (int s = 64; s > 0; s >>= 1) {
            if (d < s) red[d] = fmaxf(red[d], red[d + s]);
            __syncthreads();
        }
        float gmax = red[0];
        __syncthreads();
        wcomb[c * 128 + d] = expf(g - gmax) * w;
        __syncthreads();
    }
}

// scores[n,c] = dot(concepts[n,:], wcomb[c,:]) + lens_b[c]   (one wave/node)
__global__ void k_scores(const float* __restrict__ concepts, const float* __restrict__ wcomb,
                         const float* __restrict__ lensB, float* __restrict__ scores, int n) {
    int node = blockIdx.x * 8 + (threadIdx.x >> 5);
    int lane = threadIdx.x & 31;
    if (node >= n) return;
    float4 v = ((const float4*)(concepts + (size_t)node * 128))[lane];
    for (int c = 0; c < 10; ++c) {
        float4 w = ((const float4*)(wcomb + (size_t)c * 128))[lane];
        float p = v.x * w.x + v.y * w.y + v.z * w.z + v.w * w.w;
        p = wave_sum32(p);
        if (lane == 0) scores[(size_t)node * 10 + c] = p + lensB[c];
    }
}

// Per-class log-sum-exp over all nodes (one block per class)
__global__ void k_logz(const float* __restrict__ scores, float* __restrict__ logZ, int n) {
    __shared__ float red[256];
    int c = blockIdx.x, t = threadIdx.x;
    float mx = -3.4e38f;
    for (int i = t; i < n; i += 256) mx = fmaxf(mx, scores[(size_t)i * 10 + c]);
    red[t] = mx;
    __syncthreads();
    for (int s = 128; s > 0; s >>= 1) {
        if (t < s) red[t] = fmaxf(red[t], red[t + s]);
        __syncthreads();
    }
    float smax = red[0];
    __syncthreads();
    float sum = 0.0f;
    for (int i = t; i < n; i += 256) sum += expf(scores[(size_t)i * 10 + c] - smax);
    red[t] = sum;
    __syncthreads();
    for (int s = 128; s > 0; s >>= 1) {
        if (t < s) red[t] += red[t + s];
        __syncthreads();
    }
    if (t == 0) logZ[c] = smax + logf(red[0]);
}

__global__ void k_logprobs(const float* __restrict__ scores, const float* __restrict__ logZ,
                           float* __restrict__ out, int n) {
    int i = blockIdx.x * blockDim.x + threadIdx.x;
    if (i < n * 10) out[i] = scores[i] - logZ[i % 10];
}

// ---------------------------------------------------------------------------
// Launch
// ---------------------------------------------------------------------------
extern "C" void kernel_launch(void* const* d_in, const int* in_sizes, int n_in,
                              void* d_out, int out_size, void* d_ws, size_t ws_size,
                              hipStream_t stream) {
    const float* x     = (const float*)d_in[0];
    const int*   ei    = (const int*)d_in[1];
    const float* W[4]  = {(const float*)d_in[2], (const float*)d_in[4],
                          (const float*)d_in[6], (const float*)d_in[8]};
    const float* b[4]  = {(const float*)d_in[3], (const float*)d_in[5],
                          (const float*)d_in[7], (const float*)d_in[9]};
    const float* lensW = (const float*)d_in[10];
    const float* lensB = (const float*)d_in[11];
    float* out = (float*)d_out;

    const int N = in_sizes[0] / 128;
    const int E = in_sizes[1] / 2;
    const int* src = ei;
    const int* dst = ei + E;

    // Carve workspace (aligned 256B)
    char* ws = (char*)d_ws;
    size_t off = 0;
    auto carve = [&](size_t bytes) -> void* {
        void* p = ws + off;
        off = (off + bytes + 255) & ~(size_t)255;
        return p;
    };
    float* dis    = (float*)carve((size_t)N * 4);          // degree -> rsqrt
    float* B0     = (float*)carve((size_t)N * 128 * 4);    // h / agg (ping)
    float* B1     = (float*)carve((size_t)N * 128 * 4);    // m = hW (pong)
    float* scores = (float*)carve((size_t)N * 10 * 4);
    float* wcomb  = (float*)carve(10 * 128 * 4);
    float* logZ   = (float*)carve(10 * 4);
    (void)ws_size; (void)n_in; (void)out_size;

    // --- degree normalization: dis = rsqrt(deg), deg includes self loop ---
    k_fill1<<<(N + 255) / 256, 256, 0, stream>>>(dis, N);
    k_deg<<<(E + 255) / 256, 256, 0, stream>>>(dst, dis, E);
    k_rsqrt<<<(N + 255) / 256, 256, 0, stream>>>(dis, N);

    // --- 4 GCN layers ---
    const int gemmBlocks = (N + NODES_PER_GEMM_BLOCK - 1) / NODES_PER_GEMM_BLOCK;
    const int elemBlocks = (N * 128 + 255) / 256;
    const int edgeBlocks = (E * 32 + 255) / 256;   // one wave per edge

    const float* A = x;
    for (int l = 0; l < 4; ++l) {
        k_gemm128<<<gemmBlocks, 256, 0, stream>>>(A, W[l], B1, N);
        k_selfloop<<<elemBlocks, 256, 0, stream>>>(B1, dis, B0, N);
        k_edges<<<edgeBlocks, 256, 0, stream>>>(src, dst, dis, B1, B0, E);
        k_bias_lrelu<<<elemBlocks, 256, 0, stream>>>(B0, b[l], N);
        A = B0;
    }

    // --- concepts (first N*128 of d_out) ---
    k_concepts<<<(N + 7) / 8, 256, 0, stream>>>(B0, out, N);

    // --- entropy-lens head ---
    k_wcomb<<<1, 128, 0, stream>>>(lensW, wcomb);
    k_scores<<<(N + 7) / 8, 256, 0, stream>>>(out, wcomb, lensB, scores, N);
    k_logz<<<10, 256, 0, stream>>>(scores, logZ, N);
    k_logprobs<<<(N * 10 + 255) / 256, 256, 0, stream>>>(
        scores, logZ, out + (size_t)N * 128, N);
}
